// Attention_2439541424406
// MI455X (gfx1250) — compile-verified, matrix operand order checked by reading, not run on hardware
//
#include <hip/hip_runtime.h>
#include <hip/hip_bf16.h>

// B=2, S=2048, D=1024, H=16, DH=64  ->  M = B*S = 4096, BH = 32
// Workspace (f16): q[32][2048][64], k[32][2048][64], vT[32][64][2048], o[32][2048][64]
// total = 32 MB of d_ws.

typedef __attribute__((ext_vector_type(16))) _Float16 v16h;
typedef __attribute__((ext_vector_type(8)))  _Float16 v8h;
typedef __attribute__((ext_vector_type(8)))  float    v8f;
typedef __attribute__((ext_vector_type(4)))  float    v4f;
typedef __attribute__((ext_vector_type(4)))  unsigned int u32x4;
typedef __attribute__((ext_vector_type(8)))  int      i32x8;
typedef __attribute__((ext_vector_type(4)))  int      i32x4;

#define S_LEN 2048
#define D_DIM 1024
#define NH    16
#define DH    64
#define BH    32
#define MROWS 4096

__device__ __forceinline__ v16h load_frag(const _Float16* p) {
    // p 16B aligned; fragment = 8 contiguous halves at p, 8 more at p+16
    v8h lo = *(const v8h*)(p);
    v8h hi = *(const v8h*)(p + 16);
    return __builtin_shufflevector(lo, hi, 0,1,2,3,4,5,6,7,8,9,10,11,12,13,14,15);
}

__device__ __forceinline__ v8f wmma16(v16h a, v16h b, v8f c) {
    return __builtin_amdgcn_wmma_f32_16x16x32_f16(false, a, false, b, (short)0, c, false, false);
}

// 2D Tensor-Data-Mover load: tile (tile_d1 rows x tile_d0 elems) of a 2-byte-element
// tensor (tensor_d1 x tensor_d0, row stride stride0 elems) -> LDS at lds_off.
// D# layout per CDNA5 ISA (group0 128b, group1 256b; 2D tensor so
// groups 2/3 (and the trailing group of the 6-arg builtin) are zero).
__device__ __forceinline__ void tdm_load_2d(unsigned lds_off, const void* gptr,
                                            unsigned tensor_d0, unsigned tensor_d1,
                                            unsigned tile_d0, unsigned tile_d1,
                                            unsigned stride0) {
    unsigned long long ga = (unsigned long long)(size_t)gptr;
    u32x4 g0;
    g0[0] = 1u;                                     // count=1, user mode, no gather
    g0[1] = lds_off;                                // lds_addr (bytes)
    g0[2] = (unsigned)(ga & 0xFFFFFFFFu);           // global_addr[31:0]
    g0[3] = (unsigned)((ga >> 32) & 0x01FFFFFFu)    // global_addr[56:32]
          | (2u << 30);                             // type = 2 ("image")
    i32x8 g1;
    g1[0] = (int)(1u << 16);                        // workgroup_mask=0, data_size=1 (2B)
    g1[1] = (int)((tensor_d0 & 0xFFFFu) << 16);     // tensor_dim0[15:0]
    g1[2] = (int)(((tensor_d0 >> 16) & 0xFFFFu) | ((tensor_d1 & 0xFFFFu) << 16));
    g1[3] = (int)(((tensor_d1 >> 16) & 0xFFFFu) | ((tile_d0 & 0xFFFFu) << 16));
    g1[4] = (int)(tile_d1 & 0xFFFFu);               // tile_dim1; tile_dim2=0
    g1[5] = (int)stride0;                           // tensor_dim0_stride[31:0]
    g1[6] = 0;                                      // stride0 hi / stride1 lo
    g1[7] = 0;
    i32x4 z4 = (i32x4)0;
    i32x8 z8 = (i32x8)0;
    __builtin_amdgcn_tensor_load_to_lds(g0, g1, z4, z4, z8, 0);
}

// ---------------------------------------------------------------------------
// Kernel 1: QKV = X[4096,1024] * Wqkv[1024,3072] -> scatter f16 Q/K/Vt
// block 64(M) x 256(N), 8 waves = 2x4, wave tile 32x64 (2x4 fragments)
// ---------------------------------------------------------------------------
__global__ __launch_bounds__(256) void qkv_kernel(
    const float* __restrict__ X, const float* __restrict__ Wqkv,
    _Float16* __restrict__ qo, _Float16* __restrict__ ko, _Float16* __restrict__ vT)
{
    __shared__ alignas(16) _Float16 As[64][32];     // A-frag loads contiguous
    __shared__ alignas(16) _Float16 BsT[256][40];   // transposed -> contiguous B-frags

    const int tid  = threadIdx.x;
    const int lane = tid & 31;
    const int wid  = tid >> 5;
    const int ln   = lane & 15;
    const int hsel = (lane < 16) ? 0 : 8;
    const int m0   = blockIdx.x * 64;
    const int n0   = blockIdx.y * 256;
    const int wm   = wid >> 2, wn = wid & 3;

    const int ai  = tid >> 2,  aj0 = (tid & 3) * 8;   // A: 8 elems/thread
    const int bk  = tid >> 3,  bn0 = (tid & 7) * 32;  // B: 32 elems/thread

    v8f c[2][4];
    #pragma unroll
    for (int i = 0; i < 2; ++i)
    #pragma unroll
    for (int j = 0; j < 4; ++j) c[i][j] = 0.0f;

    for (int k0 = 0; k0 < D_DIM; k0 += 32) {
        __syncthreads();
        {   const float* ap = X + (size_t)(m0 + ai) * D_DIM + k0 + aj0;
            v4f x0 = *(const v4f*)ap, x1 = *(const v4f*)(ap + 4);
            v8h pk;
            #pragma unroll
            for (int u = 0; u < 4; ++u) { pk[u] = (_Float16)x0[u]; pk[4+u] = (_Float16)x1[u]; }
            *(v8h*)&As[ai][aj0] = pk;
        }
        {   const float* bp = Wqkv + (size_t)(k0 + bk) * (3 * D_DIM) + n0 + bn0;
            #pragma unroll
            for (int t8 = 0; t8 < 8; ++t8) {
                v4f xv = *(const v4f*)(bp + t8 * 4);
                #pragma unroll
                for (int u = 0; u < 4; ++u) BsT[bn0 + t8*4 + u][bk] = (_Float16)xv[u];
            }
        }
        __syncthreads();

        v16h a0 = load_frag(&As[wm * 32      + ln][hsel]);
        v16h a1 = load_frag(&As[wm * 32 + 16 + ln][hsel]);
        #pragma unroll
        for (int tj = 0; tj < 4; ++tj) {
            v16h b = load_frag(&BsT[wn * 64 + tj * 16 + ln][hsel]);
            c[0][tj] = wmma16(a0, b, c[0][tj]);
            c[1][tj] = wmma16(a1, b, c[1][tj]);
        }
    }

    // scatter: which/h uniform per 16-wide fragment column; V rows contiguous in s
    #pragma unroll
    for (int tj = 0; tj < 4; ++tj) {
        const int colbase = n0 + wn * 64 + tj * 16;
        const int which = colbase >> 10;
        const int h     = (colbase & 1023) >> 6;
        const int dh    = (colbase & 63) + ln;
        #pragma unroll
        for (int ti = 0; ti < 2; ++ti) {
            const int rowbase = m0 + wm * 32 + ti * 16 + hsel;
            const int b_ = rowbase >> 11, sbase = rowbase & 2047;
            const int bh = b_ * NH + h;
            if (which == 2) {               // vT[bh][dh][s]: pack 8 contiguous s
                v8h pk;
                #pragma unroll
                for (int r = 0; r < 8; ++r) pk[r] = (_Float16)c[ti][tj][r];
                *(v8h*)&vT[((size_t)bh * DH + dh) * S_LEN + sbase] = pk;
            } else {
                _Float16* dst = (which == 0) ? qo : ko;
                const float sc = (which == 0) ? 0.125f : 1.0f;  // fold 1/sqrt(DH) into Q
                #pragma unroll
                for (int r = 0; r < 8; ++r)
                    dst[((size_t)bh * S_LEN + sbase + r) * DH + dh] =
                        (_Float16)(c[ti][tj][r] * sc);
            }
        }
    }
}

// ---------------------------------------------------------------------------
// Kernel 2: flash attention, one wave per (bh, 16-row Q tile), KV step 32.
// K/V tiles streamed by the Tensor Data Mover, double-buffered in LDS.
// ---------------------------------------------------------------------------
__global__ __launch_bounds__(256) void attn_kernel(
    const _Float16* __restrict__ q, const _Float16* __restrict__ k,
    const _Float16* __restrict__ vT, _Float16* __restrict__ o)
{
    __shared__ alignas(16) _Float16 Kt[8][2][32][64];   // 64 KB (per-wave dbl buffer)
    __shared__ alignas(16) _Float16 Vt[8][2][64][32];   // 64 KB
    __shared__ alignas(16) _Float16 Pbuf[8][16][32];    //  8 KB (C-frag -> A-frag)

    const int lane = threadIdx.x & 31;
    const int ln   = lane & 15;
    const int hsel = (lane < 16) ? 0 : 8;
    const int wid  = __builtin_amdgcn_readfirstlane((int)(threadIdx.x >> 5));
    const int w    = blockIdx.x * 8 + wid;
    const int bh   = w >> 7;                 // / (S/16)
    const int qt   = w & 127;

    const _Float16* qrow = q + ((size_t)bh * S_LEN + qt * 16 + ln) * DH + hsel;
    v16h qa0 = load_frag(qrow);
    v16h qa1 = load_frag(qrow + 32);

    const _Float16* kbase = k  + (size_t)bh * S_LEN * DH;
    const _Float16* vbase = vT + (size_t)bh * DH * S_LEN;

    // prime buffer 0 with KV rows [0,32)
    tdm_load_2d((unsigned)(size_t)&Kt[wid][0][0][0], kbase, DH, S_LEN, DH, 32, DH);
    tdm_load_2d((unsigned)(size_t)&Vt[wid][0][0][0], vbase, S_LEN, DH, 32, DH, S_LEN);

    v8f o0 = 0.0f, o1 = 0.0f, o2 = 0.0f, o3 = 0.0f;
    float mrow[8], lrow[8];
    #pragma unroll
    for (int r = 0; r < 8; ++r) { mrow[r] = -1e30f; lrow[r] = 0.0f; }

    for (int j = 0; j < S_LEN; j += 32) {
        const int buf = (j >> 5) & 1;
        if (j + 32 < S_LEN) {   // issue next pair, then retire current pair
            tdm_load_2d((unsigned)(size_t)&Kt[wid][buf ^ 1][0][0],
                        kbase + (size_t)(j + 32) * DH, DH, S_LEN, DH, 32, DH);
            tdm_load_2d((unsigned)(size_t)&Vt[wid][buf ^ 1][0][0],
                        vbase + (j + 32), S_LEN, DH, 32, DH, S_LEN);
            __builtin_amdgcn_s_wait_tensorcnt(2);
        } else {
            __builtin_amdgcn_s_wait_tensorcnt(0);
        }

        // scores = Q(16x64) * K^T(64x16) for two 16-col KV tiles
        v8f c0 = 0.0f, c1 = 0.0f;
        c0 = wmma16(qa0, load_frag(&Kt[wid][buf][ln     ][hsel]),      c0);
        c0 = wmma16(qa1, load_frag(&Kt[wid][buf][ln     ][32 + hsel]), c0);
        c1 = wmma16(qa0, load_frag(&Kt[wid][buf][16 + ln][hsel]),      c1);
        c1 = wmma16(qa1, load_frag(&Kt[wid][buf][16 + ln][32 + hsel]), c1);

        // online softmax over the 32 new columns (row groups = 16 lanes)
        #pragma unroll
        for (int r = 0; r < 8; ++r) {
            float x0 = c0[r], x1 = c1[r];
            float mx = fmaxf(x0, x1);
            mx = fmaxf(mx, __shfl_xor(mx, 1, 32));
            mx = fmaxf(mx, __shfl_xor(mx, 2, 32));
            mx = fmaxf(mx, __shfl_xor(mx, 4, 32));
            mx = fmaxf(mx, __shfl_xor(mx, 8, 32));
            float mn = fmaxf(mrow[r], mx);
            float sc = __expf(mrow[r] - mn);
            float p0 = __expf(x0 - mn);
            float p1 = __expf(x1 - mn);
            float sm = p0 + p1;
            sm += __shfl_xor(sm, 1, 32);
            sm += __shfl_xor(sm, 2, 32);
            sm += __shfl_xor(sm, 4, 32);
            sm += __shfl_xor(sm, 8, 32);
            lrow[r] = lrow[r] * sc + sm;
            mrow[r] = mn;
            o0[r] *= sc; o1[r] *= sc; o2[r] *= sc; o3[r] *= sc;
            Pbuf[wid][r + hsel][ln]      = (_Float16)p0;
            Pbuf[wid][r + hsel][16 + ln] = (_Float16)p1;
        }

        // P(16x32) as A-fragment via LDS round-trip (DS in-order per wave)
        v16h pa = load_frag(&Pbuf[wid][ln][hsel]);

        // O += P * V(32x64): 4 dh tiles from the transposed V tile
        o0 = wmma16(pa, load_frag(&Vt[wid][buf][     ln][hsel]), o0);
        o1 = wmma16(pa, load_frag(&Vt[wid][buf][16 + ln][hsel]), o1);
        o2 = wmma16(pa, load_frag(&Vt[wid][buf][32 + ln][hsel]), o2);
        o3 = wmma16(pa, load_frag(&Vt[wid][buf][48 + ln][hsel]), o3);
    }

    _Float16* op = o + ((size_t)bh * S_LEN + qt * 16) * DH;
    #pragma unroll
    for (int r = 0; r < 8; ++r) {
        float inv = 1.0f / lrow[r];
        int m = r + hsel;
        op[(size_t)m * DH +      ln] = (_Float16)(o0[r] * inv);
        op[(size_t)m * DH + 16 + ln] = (_Float16)(o1[r] * inv);
        op[(size_t)m * DH + 32 + ln] = (_Float16)(o2[r] * inv);
        op[(size_t)m * DH + 48 + ln] = (_Float16)(o3[r] * inv);
    }
}

// ---------------------------------------------------------------------------
// Kernel 3: out = O[4096,1024] * Wout[1024,1024] + bias (fp32 out)
// block 64(M) x 256(N), wave tile 32x64; A gathered from [b][h][s][dh]
// ---------------------------------------------------------------------------
__global__ __launch_bounds__(256) void out_kernel(
    const _Float16* __restrict__ O, const float* __restrict__ Wout,
    const float* __restrict__ bias, float* __restrict__ out)
{
    __shared__ alignas(16) _Float16 As[64][32];
    __shared__ alignas(16) _Float16 BsT[256][40];

    const int tid  = threadIdx.x;
    const int lane = tid & 31;
    const int wid  = tid >> 5;
    const int ln   = lane & 15;
    const int hsel = (lane < 16) ? 0 : 8;
    const int m0   = blockIdx.x * 64;
    const int n0   = blockIdx.y * 256;
    const int wm   = wid >> 2, wn = wid & 3;

    const int ai  = tid >> 2,  aj0 = (tid & 3) * 8;
    const int bk  = tid >> 3,  bn0 = (tid & 7) * 32;

    v8f c[2][4];
    #pragma unroll
    for (int i = 0; i < 2; ++i)
    #pragma unroll
    for (int j = 0; j < 4; ++j) c[i][j] = 0.0f;

    const int arow = m0 + ai;
    const int ab = arow >> 11, as = arow & 2047;

    for (int k0 = 0; k0 < D_DIM; k0 += 32) {
        __syncthreads();
        {   // gather [b][h][s][dh]; 32-wide K chunk stays inside one 64-wide head
            int kc = k0 + aj0;
            int h = kc >> 6, dh0 = kc & 63;
            const _Float16* ap = O + ((size_t)(ab * NH + h) * S_LEN + as) * DH + dh0;
            *(v8h*)&As[ai][aj0] = *(const v8h*)ap;
        }
        {   const float* bp = Wout + (size_t)(k0 + bk) * D_DIM + n0 + bn0;
            #pragma unroll
            for (int t8 = 0; t8 < 8; ++t8) {
                v4f xv = *(const v4f*)(bp + t8 * 4);
                #pragma unroll
                for (int u = 0; u < 4; ++u) BsT[bn0 + t8*4 + u][bk] = (_Float16)xv[u];
            }
        }
        __syncthreads();

        v16h a0 = load_frag(&As[wm * 32      + ln][hsel]);
        v16h a1 = load_frag(&As[wm * 32 + 16 + ln][hsel]);
        #pragma unroll
        for (int tj = 0; tj < 4; ++tj) {
            v16h b = load_frag(&BsT[wn * 64 + tj * 16 + ln][hsel]);
            c[0][tj] = wmma16(a0, b, c[0][tj]);
            c[1][tj] = wmma16(a1, b, c[1][tj]);
        }
    }

    #pragma unroll
    for (int tj = 0; tj < 4; ++tj) {
        int col = n0 + wn * 64 + tj * 16 + ln;
        float bv = bias[col];
        #pragma unroll
        for (int ti = 0; ti < 2; ++ti)
        #pragma unroll
        for (int r = 0; r < 8; ++r) {
            int row = m0 + wm * 32 + ti * 16 + r + hsel;
            out[(size_t)row * D_DIM + col] = c[ti][tj][r] + bv;
        }
    }
}

extern "C" void kernel_launch(void* const* d_in, const int* in_sizes, int n_in,
                              void* d_out, int out_size, void* d_ws, size_t ws_size,
                              hipStream_t stream) {
    const float* x     = (const float*)d_in[0];   // [2,2048,1024]
    const float* w_qkv = (const float*)d_in[1];   // [1024,3072]
    const float* w_out = (const float*)d_in[2];   // [1024,1024]
    const float* b_out = (const float*)d_in[3];   // [1024]
    float* out = (float*)d_out;                   // [2,2048,1024]

    const size_t SEG = (size_t)BH * S_LEN * DH;   // 4,194,304 halves = 8 MB
    _Float16* q_ws  = (_Float16*)d_ws;
    _Float16* k_ws  = q_ws + SEG;
    _Float16* vT_ws = k_ws + SEG;
    _Float16* o_ws  = vT_ws + SEG;                // needs 32 MB total

    qkv_kernel<<<dim3(MROWS / 64, (3 * D_DIM) / 256), 256, 0, stream>>>(
        x, w_qkv, q_ws, k_ws, vT_ws);
    attn_kernel<<<dim3((BH * (S_LEN / 16)) / 8), 256, 0, stream>>>(
        q_ws, k_ws, vT_ws, o_ws);
    out_kernel<<<dim3(MROWS / 64, D_DIM / 256), 256, 0, stream>>>(
        o_ws, w_out, b_out, out);
}
